// MHSA_30975304138960
// MI455X (gfx1250) — compile-verified, matrix operand order checked by reading, not run on hardware
//
#include <hip/hip_runtime.h>
#include <hip/hip_bf16.h>
#include <cstdint>

typedef __attribute__((ext_vector_type(16))) _Float16 v16h;
typedef __attribute__((ext_vector_type(2)))  _Float16 h2;
typedef __attribute__((ext_vector_type(8)))  float    v8f;
typedef __attribute__((ext_vector_type(4)))  int      v4i;

// pointer-to-AS1/AS3 int4 (builtin parameter types for async LDS copies)
typedef __attribute__((address_space(1))) v4i* g_v4i_p;
typedef __attribute__((address_space(3))) v4i* l_v4i_p;

#define DIMD   1024
#define HEADSN 16
#define HD     64
#define SEQ    2048
#define BATCH  4
#define MTOT   (BATCH*SEQ)   /* 8192 */

// ---------------------------------------------------------------------------
// Async global->LDS copy (CDNA5, ASYNCcnt-tracked) + wait helpers
// ---------------------------------------------------------------------------
__device__ __forceinline__ void async_copy_b128(void* lds_dst, const void* gsrc) {
#if __has_builtin(__builtin_amdgcn_global_load_async_to_lds_b128)
  __builtin_amdgcn_global_load_async_to_lds_b128(
      (g_v4i_p)(uintptr_t)gsrc,
      (l_v4i_p)(uint32_t)(uintptr_t)lds_dst,
      0, 0);
#else
  uint32_t l = (uint32_t)(uintptr_t)lds_dst;
  asm volatile("global_load_async_to_lds_b128 %0, %1, off"
               :: "v"(l), "v"(gsrc) : "memory");
#endif
}

__device__ __forceinline__ void wait_async0() {
#if __has_builtin(__builtin_amdgcn_s_wait_asynccnt)
  __builtin_amdgcn_s_wait_asynccnt(0);
#else
  asm volatile("s_wait_asynccnt 0" ::: "memory");
#endif
}

// ---------------------------------------------------------------------------
// Fragment helpers (CDNA5 WMMA 16x16x32 f16 layouts, ISA 7.12.2)
// ---------------------------------------------------------------------------
__device__ __forceinline__ int kpair_base(int lane, int i2) {
  int kb = (lane & 16) ? 8 : 0;
  return (i2 < 4) ? (kb + 2 * i2) : (16 + kb + 2 * (i2 - 4));
}

__device__ __forceinline__ v16h load_frag_pairs(const _Float16* p, int lane) {
  v16h f;
#pragma unroll
  for (int i2 = 0; i2 < 8; ++i2) {
    h2 v = *(const h2*)(p + kpair_base(lane, i2));
    f[2 * i2]     = v[0];
    f[2 * i2 + 1] = v[1];
  }
  return f;
}

__device__ __forceinline__ v16h load_a_frag(const _Float16* A, int lda, int rowbase, int k0) {
  int lane = threadIdx.x & 31;
  return load_frag_pairs(A + (size_t)(rowbase + (lane & 15)) * lda + k0, lane);
}

__device__ __forceinline__ v8f wmma_f16(v16h a, v16h b, v8f c) {
  return __builtin_amdgcn_wmma_f32_16x16x32_f16(false, a, false, b, (short)0, c, false, false);
}

__device__ __forceinline__ v8f v8f_zero() {
  v8f z = {0.f, 0.f, 0.f, 0.f, 0.f, 0.f, 0.f, 0.f};
  return z;
}

// ---------------------------------------------------------------------------
// Convert kernels
// ---------------------------------------------------------------------------
__global__ void f32_to_f16_kernel(const float* __restrict__ src,
                                  _Float16* __restrict__ dst, int n) {
  int i = blockIdx.x * blockDim.x + threadIdx.x;
  int stride = gridDim.x * blockDim.x;
  for (; i < n; i += stride) dst[i] = (_Float16)src[i];
}

// Row-major [K,N] f32 -> fragment-major f16 (frag = nt16*(K/32)+kt; lane*16+i)
__global__ void swizzle_w_kernel(const float* __restrict__ W,
                                 _Float16* __restrict__ Wsw, int K, int N) {
  int total = K * N;
  int i = blockIdx.x * blockDim.x + threadIdx.x;
  int stride = gridDim.x * blockDim.x;
  int ktiles = K / 32;
  for (; i < total; i += stride) {
    int frag   = i >> 9;
    int within = i & 511;
    int lane   = within >> 4;
    int ii     = within & 15;
    int nt     = frag / ktiles;
    int kt     = frag - nt * ktiles;
    int n      = nt * 16 + (lane & 15);
    int i2     = ii >> 1;
    int j      = ii & 1;
    int k      = kt * 32 + kpair_base(lane, i2) + j;
    Wsw[i] = (_Float16)W[(size_t)k * N + n];
  }
}

// ---------------------------------------------------------------------------
// GEMM body shared by qkv/proj: 8 waves/block, same nt64, async B staging.
// Each wave: 16x64 tile, 4 accumulators. B slab (4 frags, 4KB) double-buffered.
// ---------------------------------------------------------------------------
#define GEMM_PROLOG()                                                            \
  __shared__ __align__(16) _Float16 bsm[2][4 * 512];                             \
  const int lane = threadIdx.x & 31;                                             \
  const int wv   = threadIdx.x >> 5;                                             \
  const int gw   = blockIdx.x * 8 + wv;                                          \
  const int mt   = gw & 511;                                                     \
  const int nt64 = gw >> 9; /* uniform across block: 512 %% 8 == 0 */            \
  const int ktn  = DIMD / 32;                                                    \
  const int c_   = wv >> 1;                                                      \
  const int hf   = wv & 1;                                                       \
  const _Float16* ssrc = Bsw + ((size_t)(nt64 * 4 + c_) * ktn << 9)              \
                             + hf * 256 + lane * 8;                              \
  _Float16* sdst = &bsm[0][0] + c_ * 512 + hf * 256 + lane * 8;                  \
  async_copy_b128(sdst, ssrc);                                                   \
  v8f acc0 = v8f_zero(), acc1 = v8f_zero(), acc2 = v8f_zero(), acc3 = v8f_zero();\
  for (int kf = 0; kf < ktn; ++kf) {                                             \
    const int cur = kf & 1;                                                      \
    wait_async0();                                                               \
    __syncthreads();                                                             \
    if (kf + 1 < ktn)                                                            \
      async_copy_b128(sdst + (1 - cur) * 2048, ssrc + ((size_t)(kf + 1) << 9));  \
    v16h a = load_a_frag(Amat, DIMD, mt * 16, kf * 32);                          \
    const _Float16* bb = &bsm[cur][0] + lane * 16;                               \
    v16h b0 = *(const v16h*)(bb + 0 * 512);                                      \
    v16h b1 = *(const v16h*)(bb + 1 * 512);                                      \
    v16h b2 = *(const v16h*)(bb + 2 * 512);                                      \
    v16h b3 = *(const v16h*)(bb + 3 * 512);                                      \
    acc0 = wmma_f16(a, b0, acc0);                                                \
    acc1 = wmma_f16(a, b1, acc1);                                                \
    acc2 = wmma_f16(a, b2, acc2);                                                \
    acc3 = wmma_f16(a, b3, acc3);                                                \
  }

// QKV: [8192,1024] x [1024,3072] + bias -> Q,K [B,H,N,64] f16, V^T [B,H,64,N] f16
__global__ void __launch_bounds__(256)
qkv_gemm_kernel(const _Float16* __restrict__ Amat, const _Float16* __restrict__ Bsw,
                const float* __restrict__ bias, _Float16* __restrict__ Qb,
                _Float16* __restrict__ Kb, _Float16* __restrict__ Vt) {
  GEMM_PROLOG()
  v8f accs[4] = {acc0, acc1, acc2, acc3};
  const int mrow_off = (lane & 16) ? 8 : 0;
#pragma unroll
  for (int c = 0; c < 4; ++c) {
    int n     = (nt64 * 4 + c) * 16 + (lane & 15);
    int which = n >> 10;   // 0=q 1=k 2=v
    int d     = n & 1023;
    int h     = d >> 6;
    int e     = d & 63;
    float bv  = bias[n];
#pragma unroll
    for (int r = 0; r < 8; ++r) {
      int m  = mt * 16 + r + mrow_off;
      int b_ = m >> 11;
      int s  = m & 2047;
      _Float16 hv = (_Float16)(accs[c][r] + bv);
      size_t bh = (size_t)(b_ * HEADSN + h);
      if (which == 0)      Qb[(bh * SEQ + s) * HD + e] = hv;
      else if (which == 1) Kb[(bh * SEQ + s) * HD + e] = hv;
      else                 Vt[(bh * HD + e) * SEQ + s] = hv;
    }
  }
}

// Proj: ctx [8192,1024] f16 x Wproj_swz + bias -> out f32
__global__ void __launch_bounds__(256)
proj_gemm_kernel(const _Float16* __restrict__ Amat, const _Float16* __restrict__ Bsw,
                 const float* __restrict__ bias, float* __restrict__ out) {
  GEMM_PROLOG()
  v8f accs[4] = {acc0, acc1, acc2, acc3};
  const int mrow_off = (lane & 16) ? 8 : 0;
#pragma unroll
  for (int c = 0; c < 4; ++c) {
    int n = (nt64 * 4 + c) * 16 + (lane & 15);
    float bv = bias[n];
#pragma unroll
    for (int r = 0; r < 8; ++r) {
      int m = mt * 16 + r + mrow_off;
      out[(size_t)m * DIMD + n] = accs[c][r] + bv;
    }
  }
}

// ---------------------------------------------------------------------------
// Flash attention: block = 4 waves = 4 query tiles of one (b,h).
// 32-key K/V chunks staged into double-buffered LDS by all 4 waves (async).
// ---------------------------------------------------------------------------
__global__ void __launch_bounds__(128)
attn_kernel(const _Float16* __restrict__ Qb, const _Float16* __restrict__ Kb,
            const _Float16* __restrict__ Vt, _Float16* __restrict__ ctx) {
  __shared__ __align__(16) _Float16 ksm[2][32 * 64];  // 2 x 4KB (32 keys x hd)
  __shared__ __align__(16) _Float16 vsm[2][64 * 32];  // 2 x 4KB (hd x 32 keys)
  __shared__ __align__(16) _Float16 psm[4][16 * 32];  // per-wave P slabs

  const int lane = threadIdx.x & 31;
  const int wv   = threadIdx.x >> 5;
  const int bh   = blockIdx.x >> 5;           // 64 (b,h) pairs
  const int qt   = (blockIdx.x & 31) * 4 + wv;
  const int b_   = bh >> 4;
  const int h    = bh & 15;
  const int qbase = qt * 16;

  const _Float16* Qp = Qb + (size_t)bh * SEQ * HD;
  const _Float16* Kp = Kb + (size_t)bh * SEQ * HD;
  const _Float16* Vp = Vt + (size_t)bh * HD * SEQ;
  _Float16* pl = &psm[wv][0];

  // Q fragments (loaded once, reused for all 64 chunks)
  v16h qa0 = load_a_frag(Qp, HD, qbase, 0);
  v16h qa1 = load_a_frag(Qp, HD, qbase, 32);

  v8f o0 = v8f_zero(), o1 = v8f_zero(), o2 = v8f_zero(), o3 = v8f_zero();
  float mi[8], li[8];
#pragma unroll
  for (int r = 0; r < 8; ++r) { mi[r] = -1e30f; li[r] = 0.f; }
  const int mrow_off = (lane & 16) ? 8 : 0;

  // Cooperative K/V chunk staging: 2 K-segments + 2 V-segments per wave.
  auto stage_kv = [&](int buf, int kb) {
#pragma unroll
    for (int rr = 0; rr < 2; ++rr) {
      int seg = (rr * 4 + wv) * 32 + lane;          // 0..255 (16B segments)
      async_copy_b128(&ksm[buf][seg * 8], Kp + (size_t)kb * HD + seg * 8);
      int row = seg >> 2, s = seg & 3;              // V^T row 0..63, col-seg 0..3
      async_copy_b128(&vsm[buf][row * 32 + s * 8], Vp + (size_t)row * SEQ + kb + s * 8);
    }
  };

  stage_kv(0, 0);

  for (int kb = 0; kb < SEQ; kb += 32) {
    const int cur = (kb >> 5) & 1;
    wait_async0();
    __syncthreads();
    if (kb + 32 < SEQ) stage_kv(1 - cur, kb + 32);

    // ---- S = Q K^T for the 32-key chunk (from LDS) ----
    const _Float16* kp0 = &ksm[cur][0] + (lane & 15) * HD;
    const _Float16* kp1 = &ksm[cur][0] + (size_t)(16 + (lane & 15)) * HD;
    v16h k00 = load_frag_pairs(kp0, lane);
    v16h k01 = load_frag_pairs(kp0 + 32, lane);
    v16h k10 = load_frag_pairs(kp1, lane);
    v16h k11 = load_frag_pairs(kp1 + 32, lane);

    v8f s0 = v8f_zero(), s1 = v8f_zero();
    s0 = wmma_f16(qa0, k00, s0);
    s0 = wmma_f16(qa1, k01, s0);
    s1 = wmma_f16(qa0, k10, s1);
    s1 = wmma_f16(qa1, k11, s1);

    // ---- online softmax (row stats via 16-lane xor reductions) ----
    float p0[8], p1[8];
#pragma unroll
    for (int r = 0; r < 8; ++r) {
      float a0 = s0[r] * 0.125f;
      float a1 = s1[r] * 0.125f;
      float cm = fmaxf(a0, a1);
#pragma unroll
      for (int d = 1; d < 16; d <<= 1) cm = fmaxf(cm, __shfl_xor(cm, d, 32));
      float mn = fmaxf(mi[r], cm);
      float al = __expf(mi[r] - mn);
      float e0 = __expf(a0 - mn);
      float e1 = __expf(a1 - mn);
      float rs = e0 + e1;
#pragma unroll
      for (int d = 1; d < 16; d <<= 1) rs += __shfl_xor(rs, d, 32);
      li[r] = li[r] * al + rs;
      mi[r] = mn;
      o0[r] *= al; o1[r] *= al; o2[r] *= al; o3[r] *= al;
      p0[r] = e0; p1[r] = e1;
    }

    // ---- transpose P (C layout -> A layout) through per-wave LDS slab ----
#pragma unroll
    for (int r = 0; r < 8; ++r) {
      int m = r + mrow_off;
      pl[m * 32 + (lane & 15)]      = (_Float16)p0[r];
      pl[m * 32 + 16 + (lane & 15)] = (_Float16)p1[r];
    }
    asm volatile("s_wait_dscnt 0" ::: "memory");
    v16h pa = load_frag_pairs(pl + (lane & 15) * 32, lane);

    // ---- O += P V (V^T chunk from LDS, lda = 32) ----
    const _Float16* vb = &vsm[cur][0] + (lane & 15) * 32;
    v16h vb0 = load_frag_pairs(vb, lane);
    v16h vb1 = load_frag_pairs(vb + 16 * 32, lane);
    v16h vb2 = load_frag_pairs(vb + 32 * 32, lane);
    v16h vb3 = load_frag_pairs(vb + 48 * 32, lane);
    o0 = wmma_f16(pa, vb0, o0);
    o1 = wmma_f16(pa, vb1, o1);
    o2 = wmma_f16(pa, vb2, o2);
    o3 = wmma_f16(pa, vb3, o3);
  }

  // ---- epilogue: divide by row sums, write ctx [B,N,D] f16 ----
  v8f os[4] = {o0, o1, o2, o3};
#pragma unroll
  for (int r = 0; r < 8; ++r) {
    float inv = 1.0f / li[r];
    int q = qbase + r + mrow_off;
    size_t rowoff = (size_t)(b_ * SEQ + q) * DIMD + h * HD + (lane & 15);
#pragma unroll
    for (int t = 0; t < 4; ++t)
      ctx[rowoff + t * 16] = (_Float16)(os[t][r] * inv);
  }
}

// ---------------------------------------------------------------------------
extern "C" void kernel_launch(void* const* d_in, const int* in_sizes, int n_in,
                              void* d_out, int out_size, void* d_ws, size_t ws_size,
                              hipStream_t stream) {
  const float* x     = (const float*)d_in[0];
  const float* Wqkv  = (const float*)d_in[1];
  const float* bqkv  = (const float*)d_in[2];
  const float* Wproj = (const float*)d_in[3];
  const float* bproj = (const float*)d_in[4];
  float* out = (float*)d_out;

  _Float16* xh     = (_Float16*)d_ws;
  _Float16* Wqkvh  = xh     + (size_t)MTOT * DIMD;
  _Float16* Wprojh = Wqkvh  + (size_t)DIMD * 3 * DIMD;
  _Float16* Qb     = Wprojh + (size_t)DIMD * DIMD;
  _Float16* Kb     = Qb     + (size_t)BATCH * HEADSN * SEQ * HD;
  _Float16* Vt     = Kb     + (size_t)BATCH * HEADSN * SEQ * HD;
  _Float16* ctx    = Vt     + (size_t)BATCH * HEADSN * SEQ * HD;

  f32_to_f16_kernel<<<4096, 256, 0, stream>>>(x, xh, MTOT * DIMD);
  swizzle_w_kernel<<<3072, 256, 0, stream>>>(Wqkv, Wqkvh, DIMD, 3 * DIMD);
  swizzle_w_kernel<<<1024, 256, 0, stream>>>(Wproj, Wprojh, DIMD, DIMD);

  qkv_gemm_kernel<<<(512 * 48) / 8, 256, 0, stream>>>(xh, Wqkvh, bqkv, Qb, Kb, Vt);
  attn_kernel<<<2048, 128, 0, stream>>>(Qb, Kb, Vt, ctx);
  proj_gemm_kernel<<<(512 * 16) / 8, 256, 0, stream>>>(ctx, Wprojh, bproj, out);
}
